// MultiHeadAttention_70085276336852
// MI455X (gfx1250) — matrix operand order repaired and checked by reading, not compiled, not run on hardware
//
#include <hip/hip_runtime.h>

// ---------------------------------------------------------------------------
// MHA forward for MI455X (gfx1250, wave32, WMMA).
// All matmuls (4 projections + QK^T + P*V) go through v_wmma_f32_16x16x32_bf16
// with fp32 accumulation.  The flash-attention kernel stages K/V tiles in LDS
// with double-buffered GLOBAL_LOAD_ASYNC_TO_LDS_B128 (ASYNCcnt) so the 4
// waves of a block share each 64-key tile (4x less L2 traffic) and the next
// tile's loads fly underneath the current tile's WMMAs.
// ---------------------------------------------------------------------------

#define DEVI __device__ __forceinline__

typedef __attribute__((ext_vector_type(16))) __bf16 v16bf;
typedef __attribute__((ext_vector_type(8)))  float  v8f;

constexpr int S  = 2048;
constexpr int Bz = 2;
constexpr int D  = 1024;
constexpr int H  = 16;
constexpr int DK = 64;
// scores are used in the exp2 domain: fold 1/sqrt(64) * log2(e) into the scale
constexpr float SC2 = 0.125f * 1.44269504088896340736f;

DEVI v8f wmma_bf16(v16bf a, v16bf b, v8f c) {
  // 8 args: (neg_a, A, neg_b, B, c_mod, C, reuse_a, reuse_b)
  return __builtin_amdgcn_wmma_f32_16x16x32_bf16(false, a, false, b, (short)0, c,
                                                 false, false);
}

// DPP16 lane permute within 16-lane rows (never mixes the two wave32 halves).
template <int CTRL>
DEVI float dppf(float x) {
  return __int_as_float(
      __builtin_amdgcn_update_dpp(0, __float_as_int(x), CTRL, 0xF, 0xF, true));
}
// Butterfly over 16 lanes: xor1, xor2 (quad_perm), then cross-quad via
// row_half_mirror, cross-half via row_mirror (quads are uniform by then).
DEVI float row16_max(float x) {
  x = fmaxf(x, dppf<0xB1>(x));    // quad_perm [1,0,3,2]
  x = fmaxf(x, dppf<0x4E>(x));    // quad_perm [2,3,0,1]
  x = fmaxf(x, dppf<0x141>(x));   // row_half_mirror
  x = fmaxf(x, dppf<0x140>(x));   // row_mirror
  return x;
}
DEVI float row16_sum(float x) {
  x += dppf<0xB1>(x);
  x += dppf<0x4E>(x);
  x += dppf<0x141>(x);
  x += dppf<0x140>(x);
  return x;
}

// Async global -> LDS copy of 16 bytes per lane (VGLOBAL op 98, ASYNCcnt).
DEVI void async_ld_b128(unsigned lds_off, const __bf16* gptr) {
  asm volatile("global_load_async_to_lds_b128 %0, %1, off"
               :: "v"(lds_off), "v"(gptr)
               : "memory");
}

// --- A-fragment (16x32, 16-bit): lane = row (lane&15); per-lane K range is
// [kb, kb+16) with kb = 16*(lane>=16).  (contiguous k order used by A and B here) --------
DEVI v16bf a_frag_f32(const float* __restrict__ base, int ld, int k0, int lane) {
  const int m = lane & 15, kb = (lane >> 4) << 4;
  const float* p = base + m * ld + k0 + kb;
  v16bf a;
#pragma unroll
  for (int i = 0; i < 8; ++i) {
    a[i]     = (__bf16)p[i];
    a[8 + i] = (__bf16)p[8 + i];
  }
  return a;
}

DEVI v16bf a_frag_bf16(const __bf16* base, int ld, int k0, int lane) {
  const int m = lane & 15, kb = (lane >> 4) << 4;
  const __bf16* p = base + m * ld + k0 + kb;
  v16bf a;
#pragma unroll
  for (int i = 0; i < 8; ++i) {
    a[i]     = p[i];
    a[8 + i] = p[8 + i];
  }
  return a;
}

// --- B-fragment (32x16, 16-bit): lane = column (lane&15); per-lane K range is
// [kb, kb+16) with kb = 16*(lane>=16).  B(k, n) = base[(n0+n)*ldb + k0 + k].
DEVI v16bf b_frag_wT_f32(const float* __restrict__ W, int ldw, int n0, int k0,
                         int lane) {
  const int n = lane & 15, kb = (lane >> 4) << 4;
  const float* p = W + (size_t)(n0 + n) * ldw + k0 + kb;
  v16bf b;
#pragma unroll
  for (int i = 0; i < 16; ++i) b[i] = (__bf16)p[i];
  return b;
}

DEVI v16bf b_frag_bf16(const __bf16* base, int ldb, int n0, int k0, int lane) {
  const int n = lane & 15, kb = (lane >> 4) << 4;
  const __bf16* p = base + (size_t)(n0 + n) * ldb + k0 + kb;
  v16bf b;
#pragma unroll
  for (int i = 0; i < 16; ++i) b[i] = p[i];
  return b;
}

// ---------------------------------------------------------------------------
// Projection GEMM: Y = X @ W^T + bias.  One wave owns a 16(M) x 64(N) tile.
// OUT_HEADS   : bf16 dest laid out (B,H,S,DK)   (Q and K for attention)
// OUT_HEADS_T : bf16 dest laid out (B,H,DK,S)   (V transposed for P*V B-frags)
// OUT_ROWF32  : fp32 dest laid out (M,N)        (final output projection)
// ---------------------------------------------------------------------------
enum { OUT_HEADS = 0, OUT_HEADS_T = 1, OUT_ROWF32 = 2 };

template <int IN_BF16, int OUT_MODE>
__global__ __launch_bounds__(128) void proj_kernel(const void* __restrict__ Xv,
                                                   const float* __restrict__ W,
                                                   const float* __restrict__ bias,
                                                   void* __restrict__ Yv) {
  const int lane = threadIdx.x & 31;
  const int wave = threadIdx.x >> 5;
  const int wt   = blockIdx.x * 4 + wave;   // wave-tile id in [0, 4096)
  const int tm   = wt >> 4;                 // 256 row tiles
  const int tn   = wt & 15;                 // 16 column tiles of 64
  const int row0 = tm * 16;
  const int n0   = tn * 64;

  const float*  Xf = (const float*)Xv;
  const __bf16* Xb = (const __bf16*)Xv;

  v8f acc[4] = {};

#pragma unroll 4
  for (int k0 = 0; k0 < D; k0 += 32) {
    v16bf a;
    if constexpr (IN_BF16)
      a = a_frag_bf16(Xb + (size_t)row0 * D, D, k0, lane);
    else
      a = a_frag_f32(Xf + (size_t)row0 * D, D, k0, lane);
#pragma unroll
    for (int j = 0; j < 4; ++j) {
      v16bf b = b_frag_wT_f32(W, D, n0 + j * 16, k0, lane);
      acc[j] = wmma_bf16(a, b, acc[j]);
    }
  }

  const int l16 = lane & 15, lh = lane >> 4;
#pragma unroll
  for (int j = 0; j < 4; ++j) {
    const int n = n0 + j * 16 + l16;
    const float bn = bias[n];
#pragma unroll
    for (int r = 0; r < 8; ++r) {
      const int m = row0 + r + lh * 8;      // C/D layout: row = r + 8*(lane>=16)
      const float y = acc[j][r] + bn;
      if constexpr (OUT_MODE == OUT_ROWF32) {
        ((float*)Yv)[(size_t)m * D + n] = y;
      } else {
        const int s = m >> 1, b = m & 1;    // row m = s*B + b (B = 2)
        const int h = n >> 6, dk = n & 63;  // col n = h*DK + dk (DK = 64)
        __bf16* Y = (__bf16*)Yv;
        if constexpr (OUT_MODE == OUT_HEADS)
          Y[(((size_t)b * H + h) * S + s) * DK + dk] = (__bf16)y;
        else
          Y[(((size_t)b * H + h) * DK + dk) * S + s] = (__bf16)y;
      }
    }
  }
}

// ---------------------------------------------------------------------------
// Flash attention.  Block = 4 waves = 64 query rows of one (b,h); the key axis
// is streamed in 64-key blocks.  K (64x64, key-major) and V (64x64, dk-major)
// tiles are staged in LDS via double-buffered async copies and shared by all
// 4 waves; P tiles are staged per-wave to convert C-layout -> A-fragments.
// ---------------------------------------------------------------------------
constexpr int BN   = 64;                 // keys per block
constexpr int LDT  = BN + 8;             // padded LDS row stride (144 B)
constexpr int TILE = 64 * LDT;           // elements per K or V tile
// smem layout (bf16 elements): K[2] | V[2] | P[4 waves]
constexpr int KOFF = 0;
constexpr int VOFF = 2 * TILE;
constexpr int POFF = 4 * TILE;
constexpr int SMEM_ELTS = POFF + 4 * 16 * LDT;   // ~46 KB of the 320 KB LDS

// Each of the 128 threads copies 4 x 16B chunks of K and of V for one tile.
DEVI void issue_tile_loads(unsigned kOff, unsigned vOff,
                           const __bf16* Kbh, const __bf16* Vbh,
                           int kb, int tid) {
#pragma unroll
  for (int i = 0; i < 4; ++i) {
    const int c   = i * 128 + tid;       // chunk id in [0, 512)
    const int row = c >> 3;              // 64 rows per tile
    const int col = (c & 7) * 8;         // bf16 offset within row
    const unsigned lo = (unsigned)(row * LDT + col) * 2u;
    async_ld_b128(kOff + lo, Kbh + (size_t)(kb + row) * DK + col);  // row=key
    async_ld_b128(vOff + lo, Vbh + (size_t)row * S + kb + col);     // row=dk
  }
}

__global__ __launch_bounds__(128) void fa_kernel(const __bf16* __restrict__ Q,
                                                 const __bf16* __restrict__ K,
                                                 const __bf16* __restrict__ Vt,
                                                 __bf16* __restrict__ X) {
  __shared__ __bf16 smem[SMEM_ELTS];

  const int tid  = threadIdx.x;
  const int lane = tid & 31;
  const int wave = tid >> 5;
  const int l16 = lane & 15, lh = lane >> 4;
  const int bh = blockIdx.y;                   // b*H + h
  const int b = bh / H, h = bh % H;
  const int q0 = (blockIdx.x * 4 + wave) * 16;

  const __bf16* Qbh = Q  + (size_t)bh * S * DK;
  const __bf16* Kbh = K  + (size_t)bh * S * DK;
  const __bf16* Vbh = Vt + (size_t)bh * DK * S;

  const unsigned lds_base = (unsigned)(uintptr_t)&smem[0];

  // Q fragments for DK chunks [0,32) and [32,64): loaded once, reused 32x.
  const v16bf aq0 = a_frag_bf16(Qbh + (size_t)q0 * DK, DK, 0, lane);
  const v16bf aq1 = a_frag_bf16(Qbh + (size_t)q0 * DK, DK, 32, lane);

  v8f acc[4] = {};
  float rmax[8], rsum[8];
#pragma unroll
  for (int r = 0; r < 8; ++r) { rmax[r] = -1e30f; rsum[r] = 0.0f; }

  __bf16* pw = &smem[POFF + wave * 16 * LDT];

  // prologue: start tile 0 into buffer 0
  issue_tile_loads(lds_base + KOFF * 2u, lds_base + VOFF * 2u, Kbh, Vbh, 0, tid);

  int buf = 0;
  for (int kb0 = 0; kb0 < S; kb0 += BN, buf ^= 1) {
    // kick off the next tile into the other buffer, then retire the older batch
    if (kb0 + BN < S) {
      const unsigned nb = buf ^ 1;
      issue_tile_loads(lds_base + (KOFF + nb * TILE) * 2u,
                       lds_base + (VOFF + nb * TILE) * 2u,
                       Kbh, Vbh, kb0 + BN, tid);
      asm volatile("s_wait_asynccnt 0x8" ::: "memory");  // older 8 retired
    } else {
      asm volatile("s_wait_asynccnt 0x0" ::: "memory");
    }
    __syncthreads();                       // tile[buf] visible block-wide

    const __bf16* Kt = &smem[KOFF + buf * TILE];   // (key, dk), stride LDT
    const __bf16* Vl = &smem[VOFF + buf * TILE];   // (dk, key), stride LDT

    // scores: 16 q x 64 keys = four 16x16 C tiles, K-dim = DK = 64 (2 WMMA ea)
    v8f sc[4];
#pragma unroll
    for (int t = 0; t < 4; ++t) {
      v8f c = {};
      c = wmma_bf16(aq0, b_frag_bf16(Kt, LDT, t * 16, 0, lane), c);
      c = wmma_bf16(aq1, b_frag_bf16(Kt, LDT, t * 16, 32, lane), c);
      sc[t] = c;
    }

    // online softmax (exp2 domain); lanes 0-15 carry row r, lanes 16-31 row
    // r+8, and the DPP16 butterflies reduce strictly within 16-lane rows.
#pragma unroll
    for (int r = 0; r < 8; ++r) {
      const float s0 = sc[0][r] * SC2;
      const float s1 = sc[1][r] * SC2;
      const float s2 = sc[2][r] * SC2;
      const float s3 = sc[3][r] * SC2;
      const float mx = row16_max(fmaxf(fmaxf(s0, s1), fmaxf(s2, s3)));
      const float mnew = fmaxf(rmax[r], mx);
      const float corr = __builtin_amdgcn_exp2f(rmax[r] - mnew);
      rmax[r] = mnew;
      const float p0 = __builtin_amdgcn_exp2f(s0 - mnew);
      const float p1 = __builtin_amdgcn_exp2f(s1 - mnew);
      const float p2 = __builtin_amdgcn_exp2f(s2 - mnew);
      const float p3 = __builtin_amdgcn_exp2f(s3 - mnew);
      const float ls = row16_sum((p0 + p1) + (p2 + p3));
      rsum[r] = rsum[r] * corr + ls;
#pragma unroll
      for (int j = 0; j < 4; ++j) acc[j][r] *= corr;
      // stage un-normalized P (bf16) into this wave's private LDS tile
      const int ro = (r + lh * 8) * LDT;
      pw[ro + l16]      = (__bf16)p0;
      pw[ro + 16 + l16] = (__bf16)p1;
      pw[ro + 32 + l16] = (__bf16)p2;
      pw[ro + 48 + l16] = (__bf16)p3;
    }
    asm volatile("s_wait_dscnt 0x0" ::: "memory");   // P store->load turnaround

    // reload P as two 16x32 bf16 A-fragments and do P*V from the shared tile:
    // B(k=key, n=dk) = Vl[dk*LDT + key], contiguous per lane over k.
    const v16bf ap0 = a_frag_bf16(pw, LDT, 0, lane);
    const v16bf ap1 = a_frag_bf16(pw, LDT, 32, lane);
#pragma unroll
    for (int j = 0; j < 4; ++j) {
      acc[j] = wmma_bf16(ap0, b_frag_bf16(Vl, LDT, j * 16, 0, lane), acc[j]);
      acc[j] = wmma_bf16(ap1, b_frag_bf16(Vl, LDT, j * 16, 32, lane), acc[j]);
    }
    asm volatile("s_wait_dscnt 0x0" ::: "memory");   // all DS reads landed
    __syncthreads();                       // safe to overwrite tile[buf]
  }

  // normalize and write bf16 attention output in (s, b, d) row-major layout
  float inv[8];
#pragma unroll
  for (int r = 0; r < 8; ++r) inv[r] = 1.0f / rsum[r];
#pragma unroll
  for (int j = 0; j < 4; ++j) {
    const int dk = j * 16 + l16;
#pragma unroll
    for (int r = 0; r < 8; ++r) {
      const int s = q0 + r + lh * 8;
      X[((size_t)s * Bz + b) * D + h * DK + dk] = (__bf16)(acc[j][r] * inv[r]);
    }
  }
}

// ---------------------------------------------------------------------------
extern "C" void kernel_launch(void* const* d_in, const int* in_sizes, int n_in,
                              void* d_out, int out_size, void* d_ws, size_t ws_size,
                              hipStream_t stream) {
  const float* query = (const float*)d_in[0];
  const float* key   = (const float*)d_in[1];
  const float* value = (const float*)d_in[2];
  const float* wq    = (const float*)d_in[3];
  const float* bq    = (const float*)d_in[4];
  const float* wk    = (const float*)d_in[5];
  const float* bk    = (const float*)d_in[6];
  const float* wv    = (const float*)d_in[7];
  const float* bv    = (const float*)d_in[8];
  const float* wo    = (const float*)d_in[9];
  const float* bo    = (const float*)d_in[10];
  float* out = (float*)d_out;

  // workspace: 4 bf16 buffers of S*B*D elements (8 MB each, 32 MB total)
  const size_t HE = (size_t)Bz * H * S * DK;
  __bf16* qb  = (__bf16*)d_ws;
  __bf16* kb  = qb + HE;
  __bf16* vtb = kb + HE;
  __bf16* xb  = vtb + HE;

  const dim3 pg(1024), pb(128);
  // Q, K in (B,H,S,DK); V transposed to (B,H,DK,S) for contiguous B-fragments.
  proj_kernel<0, OUT_HEADS>  <<<pg, pb, 0, stream>>>(query, wq, bq, qb);
  proj_kernel<0, OUT_HEADS>  <<<pg, pb, 0, stream>>>(key,   wk, bk, kb);
  proj_kernel<0, OUT_HEADS_T><<<pg, pb, 0, stream>>>(value, wv, bv, vtb);

  fa_kernel<<<dim3(S / 64, Bz * H), 128, 0, stream>>>(qb, kb, vtb, xb);

  proj_kernel<1, OUT_ROWF32> <<<pg, pb, 0, stream>>>(xb, wo, bo, out);
}
